// LISTACPSS_model_63453846831764
// MI455X (gfx1250) — compile-verified, matrix operand order checked by reading, not run on hardware
//
#include <hip/hip_runtime.h>
#include <cmath>

// ---------------------------------------------------------------------------
// LISTA-CPSS on MI455X (gfx1250, wave32, WMMA).
//   u_{i+1} = shrink_ss( u_i + (x - u_i A^T) W_i^T , threshold[i], PS[i] )
// B=16384, M=256, N=1024, DEPTH=16.
// Strategy: bf16x3 split GEMMs on v_wmma_f32_16x16x32_bf16 (8x the K of the
// f32 WMMA), fused per-iteration kernel (both GEMMs + exact per-row top-K
// selection + shrink), u updated in place in d_out. A/W pre-split into ws.
// x tile staged global->LDS with CDNA5 async-to-LDS (ASYNCcnt) copies.
// ---------------------------------------------------------------------------

typedef __attribute__((ext_vector_type(16))) __bf16 v16bf;
typedef __attribute__((ext_vector_type(8)))  float  v8f;
typedef unsigned int u32;
typedef __attribute__((ext_vector_type(4))) u32   u32x4;
typedef __attribute__((ext_vector_type(2))) u32   u32x2;
typedef __attribute__((ext_vector_type(4))) float f32x4;

#define DEPTH   16
#define BSZ     16384
#define MDIM    256
#define NDIM    1024
#define ROWS    32          // B-rows per workgroup
#define THREADS 256         // 8 waves of 32

// ---- bf16 helpers (bit tricks; storage is ushort, math is f32) ------------
__device__ __forceinline__ u32 bf16_rne(float f) {
  u32 u = __float_as_uint(f);
  return (u + 0x7fffu + ((u >> 16) & 1u)) >> 16;
}
__device__ __forceinline__ void bf16_split(float f, u32 &h, u32 &l) {
  h = bf16_rne(f);
  float fh = __uint_as_float(h << 16);
  l = bf16_rne(f - fh);
}

// Load one 16x32 bf16 WMMA operand fragment for this lane.
// p must point at element [row*ldk + kbase + (lane>>4)*8]; layout per ISA:
// v16bf elems 0..7 = K+0..7, elems 8..15 = K+16..23 (relative to that base).
__device__ __forceinline__ v16bf ldfrag(const unsigned short* p) {
  union { u32x4 q[2]; v16bf v; } uu;
  uu.q[0] = *(const u32x4*)(p);
  uu.q[1] = *(const u32x4*)(p + 16);
  return uu.v;
}

// bf16x3 split product accumulate: C += (ah+al) x (bh+bl)  (drop lo*lo)
__device__ __forceinline__ v8f wmma3(v16bf ah, v16bf al, v16bf bh, v16bf bl, v8f c) {
  c = __builtin_amdgcn_wmma_f32_16x16x32_bf16(false, ah, false, bh, (short)0, c, false, false);
  c = __builtin_amdgcn_wmma_f32_16x16x32_bf16(false, ah, false, bl, (short)0, c, false, false);
  c = __builtin_amdgcn_wmma_f32_16x16x32_bf16(false, al, false, bh, (short)0, c, false, false);
  return c;
}

// CDNA5 async global->LDS copy, 16B per lane, tracked on ASYNCcnt.
__device__ __forceinline__ void async_copy_b128(u32 ldsAddr, const void* gaddr) {
  asm volatile("global_load_async_to_lds_b128 %0, %1, off"
               :: "v"(ldsAddr), "v"(gaddr) : "memory");
}
__device__ __forceinline__ void wait_asynccnt0() {
  asm volatile("s_wait_asynccnt 0x0" ::: "memory");
}

// ---- one-time split of A [256x1024] and W [16x1024x256] into hi/lo bf16 ----
__global__ __launch_bounds__(256) void split_bf16_kernel(
    const float* __restrict__ src, unsigned short* __restrict__ hi,
    unsigned short* __restrict__ lo, int n4) {
  int i = blockIdx.x * blockDim.x + threadIdx.x;
  if (i >= n4) return;
  f32x4 v = ((const f32x4*)src)[i];
  u32 h0,l0,h1,l1,h2,l2,h3,l3;
  bf16_split(v.x, h0, l0); bf16_split(v.y, h1, l1);
  bf16_split(v.z, h2, l2); bf16_split(v.w, h3, l3);
  u32x2 hp; hp.x = h0 | (h1 << 16); hp.y = h2 | (h3 << 16);
  u32x2 lp; lp.x = l0 | (l1 << 16); lp.y = l2 | (l3 << 16);
  ((u32x2*)hi)[i] = hp;
  ((u32x2*)lo)[i] = lp;
}

// ---- fused per-iteration kernel -------------------------------------------
__global__ __launch_bounds__(THREADS) void lista_iter_kernel(
    const float* __restrict__ x,
    const unsigned short* __restrict__ Ahi, const unsigned short* __restrict__ Alo,
    const unsigned short* __restrict__ Whi, const unsigned short* __restrict__ Wlo,
    const float* __restrict__ thresholds,
    float* __restrict__ u,
    int iter, int Kc, int first)
{
  extern __shared__ unsigned char smem[];
  // [0, 128K): u-split (phases 0-1), later aliased as vTile f32 (phases 2-3)
  unsigned short* uHi   = (unsigned short*)smem;                 // [32][1024]
  unsigned short* uLo   = uHi + ROWS * NDIM;                     // [32][1024]
  float*          vTile = (float*)smem;                          // [32][1024]
  unsigned short* rHi   = (unsigned short*)(smem + 2u*ROWS*NDIM*2);   // [32][256]
  unsigned short* rLo   = rHi + ROWS * MDIM;                          // [32][256]
  float*          xTile = (float*)(smem + 2u*ROWS*NDIM*2 + 2u*ROWS*MDIM*2); // [32][256]
  float*          rowThr = (float*)((unsigned char*)xTile + ROWS*MDIM*4);   // [32]

  const int tid  = threadIdx.x;
  const int w    = tid >> 5;           // wave 0..7
  const int lane = tid & 31;
  const int l16  = lane & 15;
  const int hx   = lane >> 4;          // half-wave: K-phase selector
  const int mtile = w & 1;             // which 16-row half of the 32-row tile
  const int ngrp  = w >> 1;            // 0..3 column group
  const int rowBase = blockIdx.x * ROWS;

  const unsigned short* Wihi = Whi + (size_t)iter * NDIM * MDIM;
  const unsigned short* Wilo = Wlo + (size_t)iter * NDIM * MDIM;

  // ---- Kick off async staging of the x tile (contiguous 32KB) into LDS ----
  // 2048 x 16B chunks, 8 per thread; coalesced 512B per wave-instruction.
  {
    const float* xg = x + (size_t)rowBase * MDIM;
#pragma unroll
    for (int k = 0; k < ROWS * MDIM / 4 / THREADS; ++k) {   // 8 chunks
      int chunk = tid + k * THREADS;
      async_copy_b128((u32)(size_t)(xTile + chunk * 4), xg + chunk * 4);
    }
  }

  // ---- Phase 0: stage split u-tile into LDS (skip on iter 0: u == 0) ----
  if (!first) {
    for (int i4 = tid; i4 < ROWS * NDIM / 4; i4 += THREADS) {
      int elem = i4 * 4, row = elem >> 10, col = elem & (NDIM - 1);
      f32x4 v = *(const f32x4*)(u + (size_t)(rowBase + row) * NDIM + col);
      u32 h0,l0,h1,l1,h2,l2,h3,l3;
      bf16_split(v.x, h0, l0); bf16_split(v.y, h1, l1);
      bf16_split(v.z, h2, l2); bf16_split(v.w, h3, l3);
      u32x2 hp; hp.x = h0 | (h1 << 16); hp.y = h2 | (h3 << 16);
      u32x2 lp; lp.x = l0 | (l1 << 16); lp.y = l2 | (l3 << 16);
      *(u32x2*)(uHi + elem) = hp;
      *(u32x2*)(uLo + elem) = lp;
    }
  }
  wait_asynccnt0();      // our async x-chunks are in LDS; barrier publishes all
  __syncthreads();

  // ---- Phase 1: r = x - u A^T   (out [32 x 256], K = 1024) ----
  v8f c1[4] = {};
  if (!first) {
    const unsigned short* ua = uHi + (size_t)(mtile * 16 + l16) * NDIM + hx * 8;
    const unsigned short* ul = uLo + (size_t)(mtile * 16 + l16) * NDIM + hx * 8;
#pragma unroll 1
    for (int ks = 0; ks < NDIM; ks += 32) {
      v16bf ah = ldfrag(ua + ks);
      v16bf al = ldfrag(ul + ks);
#pragma unroll
      for (int t = 0; t < 4; ++t) {
        int oc = (ngrp * 4 + t) * 16 + l16;                 // output col = m
        const unsigned short* pbh = Ahi + (size_t)oc * NDIM + ks + hx * 8;
        const unsigned short* pbl = Alo + (size_t)oc * NDIM + ks + hx * 8;
        c1[t] = wmma3(ah, al, ldfrag(pbh), ldfrag(pbl), c1[t]);
      }
    }
  }
  // epilogue: r = x - acc, store split r into LDS (x comes from LDS stage)
#pragma unroll
  for (int t = 0; t < 4; ++t) {
    int colM = (ngrp * 4 + t) * 16 + l16;
#pragma unroll
    for (int j = 0; j < 8; ++j) {
      int rowM = mtile * 16 + hx * 8 + j;
      float xv = xTile[rowM * MDIM + colM];
      float r  = xv - c1[t][j];
      u32 h, l; bf16_split(r, h, l);
      rHi[rowM * MDIM + colM] = (unsigned short)h;
      rLo[rowM * MDIM + colM] = (unsigned short)l;
    }
  }
  __syncthreads();

  // ---- Phase 2: v = u + r W_i^T   (out [32 x 1024], K = 256) ----
  v8f c2[16];
  if (first) {
#pragma unroll
    for (int t = 0; t < 16; ++t)
#pragma unroll
      for (int j = 0; j < 8; ++j) c2[t][j] = 0.f;
  } else {
#pragma unroll
    for (int t = 0; t < 16; ++t) {
      int colV = (ngrp * 16 + t) * 16 + l16;
#pragma unroll
      for (int j = 0; j < 8; ++j) {
        int rowM = mtile * 16 + hx * 8 + j;
        c2[t][j] = u[(size_t)(rowBase + rowM) * NDIM + colV];   // L2 hit
      }
    }
  }
  {
    const unsigned short* ra = rHi + (size_t)(mtile * 16 + l16) * MDIM + hx * 8;
    const unsigned short* rl = rLo + (size_t)(mtile * 16 + l16) * MDIM + hx * 8;
#pragma unroll 1
    for (int ks = 0; ks < MDIM; ks += 32) {
      v16bf ah = ldfrag(ra + ks);
      v16bf al = ldfrag(rl + ks);
#pragma unroll
      for (int t = 0; t < 16; ++t) {
        int colV = (ngrp * 16 + t) * 16 + l16;                  // output col = n
        const unsigned short* pbh = Wihi + (size_t)colV * MDIM + ks + hx * 8;
        const unsigned short* pbl = Wilo + (size_t)colV * MDIM + ks + hx * 8;
        c2[t] = wmma3(ah, al, ldfrag(pbh), ldfrag(pbl), c2[t]);
      }
    }
  }
  // epilogue: spill v-tile to LDS (aliases u-split region; safe post-barrier)
#pragma unroll
  for (int t = 0; t < 16; ++t) {
    int colV = (ngrp * 16 + t) * 16 + l16;
#pragma unroll
    for (int j = 0; j < 8; ++j) {
      int rowM = mtile * 16 + hx * 8 + j;
      vTile[rowM * NDIM + colV] = c2[t][j];
    }
  }
  __syncthreads();

  // ---- Phase 3a: exact Kc-th largest |v| per row (bit-prefix search) ----
  // thresh = max T with count(|v| >= T) >= Kc  == the idx-th order statistic.
#pragma unroll 1
  for (int rr = 0; rr < ROWS / 8; ++rr) {      // 4 rows per wave
    int row = w * (ROWS / 8) + rr;
    u32 e[32];
    const u32x4* vp = (const u32x4*)(vTile + row * NDIM + lane * 32);
#pragma unroll
    for (int q = 0; q < 8; ++q) {
      u32x4 t4 = vp[q];
      e[q*4+0] = t4.x & 0x7fffffffu;
      e[q*4+1] = t4.y & 0x7fffffffu;
      e[q*4+2] = t4.z & 0x7fffffffu;
      e[q*4+3] = t4.w & 0x7fffffffu;
    }
    // row max: the result's MSB cannot exceed msb(max) -> skip dead bits
    u32 mx = 0;
#pragma unroll
    for (int j = 0; j < 32; ++j) mx = e[j] > mx ? e[j] : mx;
#pragma unroll
    for (int off = 16; off > 0; off >>= 1) {
      u32 o = (u32)__shfl_xor((int)mx, off, 32);
      mx = o > mx ? o : mx;
    }
    u32 pref = 0u;
    int bstart = (mx == 0u) ? -1 : (31 - __clz((int)mx));
#pragma unroll 1
    for (int b = bstart; b >= 0; --b) {
      u32 cand = pref | (1u << b);
      int cnt = 0;
#pragma unroll
      for (int j = 0; j < 32; ++j) cnt += (e[j] >= cand) ? 1 : 0;
#pragma unroll
      for (int off = 16; off > 0; off >>= 1) cnt += __shfl_xor(cnt, off, 32);
      if (cnt >= Kc) pref = cand;
    }
    if (lane == 0) rowThr[row] = __uint_as_float(pref);
  }
  __syncthreads();

  // ---- Phase 3b: support-selection shrink, write u in place ----
  const float tcoef = thresholds[iter];
#pragma unroll 1
  for (int idx = tid; idx < ROWS * NDIM; idx += THREADS) {
    int row = idx >> 10;
    float v  = vTile[idx];
    float rt = rowThr[row];
    float av = fabsf(v);
    bool  m  = (av >= tcoef) && (av >= rt);
    float sh = copysignf(fmaxf(av - tcoef, 0.f), v);
    u[(size_t)(rowBase + row) * NDIM + (idx & (NDIM - 1))] = m ? v : sh;
  }
}

// ---------------------------------------------------------------------------
extern "C" void kernel_launch(void* const* d_in, const int* in_sizes, int n_in,
                              void* d_out, int out_size, void* d_ws, size_t ws_size,
                              hipStream_t stream) {
  (void)in_sizes; (void)n_in; (void)out_size; (void)ws_size;
  const float* x   = (const float*)d_in[0];   // [16384, 256]
  const float* A   = (const float*)d_in[1];   // [256, 1024]
  const float* W   = (const float*)d_in[2];   // [16, 1024, 256]
  const float* thr = (const float*)d_in[3];   // [16]
  float* u = (float*)d_out;                   // [16384, 1024], evolves in place

  // workspace: split A (1MB) + split W (16MB)
  unsigned char* ws = (unsigned char*)d_ws;
  unsigned short* Ahi = (unsigned short*)ws;
  unsigned short* Alo = Ahi + (size_t)MDIM * NDIM;
  unsigned short* Whi = (unsigned short*)(ws + 2u * (size_t)MDIM * NDIM * sizeof(unsigned short));
  unsigned short* Wlo = Whi + (size_t)DEPTH * NDIM * MDIM;

  int nA4 = MDIM * NDIM / 4;
  int nW4 = DEPTH * NDIM * MDIM / 4;
  split_bf16_kernel<<<(nA4 + 255) / 256, 256, 0, stream>>>(A, Ahi, Alo, nA4);
  split_bf16_kernel<<<(nW4 + 255) / 256, 256, 0, stream>>>(W, Whi, Wlo, nW4);

  const size_t smemBytes = 2u * ROWS * NDIM * 2 /*u split / vTile*/
                         + 2u * ROWS * MDIM * 2 /*r split*/
                         + (size_t)ROWS * MDIM * 4 /*x tile*/
                         + ROWS * sizeof(float);              // 196736 B
  hipFuncSetAttribute((const void*)lista_iter_kernel,
                      hipFuncAttributeMaxDynamicSharedMemorySize, (int)smemBytes);

  for (int it = 0; it < DEPTH; ++it) {
    double p = (it + 1) * 1.2; if (p > 5.0) p = 5.0;
    int idx = (int)std::ceil((100.0 - p) * (double)NDIM / 100.0) - 1;
    int K = NDIM - idx;                       // 13..52
    lista_iter_kernel<<<BSZ / ROWS, THREADS, smemBytes, stream>>>(
        x, Ahi, Alo, Whi, Wlo, thr, u, it, K, it == 0 ? 1 : 0);
  }
}
